// Attention_41437844471833
// MI455X (gfx1250) — matrix operand order repaired and checked by reading, not compiled, not run on hardware
//
#include <hip/hip_runtime.h>
#include <hip/hip_bf16.h>

typedef __attribute__((ext_vector_type(16))) _Float16 v16h;
typedef __attribute__((ext_vector_type(8)))  _Float16 v8h;
typedef __attribute__((ext_vector_type(8)))  float    v8f;

// Load a 16-half WMMA fragment from LDS as two 16-byte ds_load_b128.
static __device__ __forceinline__ v16h ld_frag(const _Float16* p) {
    const v8h* q = (const v8h*)p;
    v8h lo = q[0], hi = q[1];
    return __builtin_shufflevector(lo, hi, 0, 1, 2, 3, 4, 5, 6, 7,
                                   8, 9, 10, 11, 12, 13, 14, 15);
}

// A-layout mapping: k within a 32-chunk -> position k (contiguous order)
static __device__ __forceinline__ int a_perm(int k) {
    return k;
}

// ---------------------------------------------------------------------------
// Generic batched GEMM: C[b][m,n] = sum_k W[m,k] * X[b][k,n] + bias[m]
// fp32 in/out, f16 WMMA with fp32 accumulation.
// grid = (N/64, M/128, batch), block = 256 (8 wave32), each wave: 16(M)x64(N).
// LDS tiles stored in fragment order -> ds_load_b128 fragment reads.
// ---------------------------------------------------------------------------
__global__ void gemm_wmma_kernel(const float* __restrict__ W,
                                 const float* __restrict__ X,
                                 const float* __restrict__ bias,
                                 float* __restrict__ C,
                                 int M, int N, int K) {
    __shared__ __align__(16) _Float16 As[128 * 32]; // As[m*32 + a_perm(k)]
    __shared__ __align__(16) _Float16 Bt[64 * 32];  // Bt[n*32 + k]
    const int tid  = threadIdx.x;
    const int wave = tid >> 5;
    const int lane = tid & 31;
    const int l16  = lane & 15;
    const int lh   = (lane >> 4) & 1;
    const int nBlock = blockIdx.x * 64;
    const int mBlock = blockIdx.y * 128;
    const int b      = blockIdx.z;
    const float* Xb = X + (size_t)b * K * N;
    float*       Cb = C + (size_t)b * M * N;

    v8f acc[4];
    for (int i = 0; i < 4; ++i)
        for (int j = 0; j < 8; ++j) acc[i][j] = 0.0f;

    for (int k0 = 0; k0 < K; k0 += 32) {
        // Stage W tile (coalesced read over k, contiguous LDS write)
        for (int idx = tid; idx < 128 * 32; idx += 256) {
            int r = idx >> 5, c = idx & 31;
            As[r * 32 + a_perm(c)] =
                (_Float16)W[(size_t)(mBlock + r) * K + (k0 + c)];
        }
        // Stage X tile transposed (coalesced read over n, Bt[n][k])
        for (int idx = tid; idx < 32 * 64; idx += 256) {
            int r = idx >> 6, c = idx & 63;
            Bt[c * 32 + r] = (_Float16)Xb[(size_t)(k0 + r) * N + (nBlock + c)];
        }
        __syncthreads();

        const v16h a = ld_frag(&As[(wave * 16 + l16) * 32 + (lh << 4)]);
        #pragma unroll
        for (int nt = 0; nt < 4; ++nt) {
            const v16h bf = ld_frag(&Bt[(nt * 16 + l16) * 32 + (lh << 4)]);
            acc[nt] = __builtin_amdgcn_wmma_f32_16x16x32_f16(
                false, a, false, bf, (short)0, acc[nt], false, false);
        }
        __syncthreads();
    }

    // C/D layout: m = r + 8*(lane/16), n = lane%16
    for (int nt = 0; nt < 4; ++nt)
        for (int r = 0; r < 8; ++r) {
            int m = mBlock + wave * 16 + r + (lh << 3);
            int n = nBlock + nt * 16 + l16;
            Cb[(size_t)m * N + n] = acc[nt][r] + bias[m];
        }
}

// ---------------------------------------------------------------------------
// Offset network: depthwise 4x4/s4 conv -> LayerNorm(64ch) -> LeakyReLU(0.2)
// -> 1x1 to (y,x) -> + ref grid -> clip -> pos[bg][s][2]
// grid = (256 spatial, 16 bg), block = 64 (one thread per channel)
// ---------------------------------------------------------------------------
__global__ void offset_kernel(const float* __restrict__ q,
                              const float* __restrict__ dw_w,
                              const float* __restrict__ dw_b,
                              const float* __restrict__ ln_g,
                              const float* __restrict__ ln_b,
                              const float* __restrict__ off_w,
                              float* __restrict__ pos) {
    const int s  = blockIdx.x;
    const int bg = blockIdx.y;
    const int c  = threadIdx.x;
    const int b = bg >> 2, g = bg & 3;
    const int i = s >> 4, j = s & 15;
    __shared__ float s1[64], s2[64];

    const float* qc = q + ((size_t)(b * 256 + g * 64 + c)) * 4096;
    float acc = dw_b[c];
    #pragma unroll
    for (int u = 0; u < 4; ++u)
        #pragma unroll
        for (int v = 0; v < 4; ++v)
            acc += qc[(i * 4 + u) * 64 + (j * 4 + v)] * dw_w[c * 16 + u * 4 + v];

    s1[c] = acc; s2[c] = acc * acc;
    __syncthreads();
    for (int off = 32; off > 0; off >>= 1) {
        if (c < off) { s1[c] += s1[c + off]; s2[c] += s2[c + off]; }
        __syncthreads();
    }
    float mu  = s1[0] * (1.0f / 64.0f);
    float var = s2[0] * (1.0f / 64.0f) - mu * mu;
    __syncthreads();

    float on = (acc - mu) * rsqrtf(var + 1e-5f) * ln_g[c] + ln_b[c];
    on = on > 0.0f ? on : 0.2f * on;

    s1[c] = on * off_w[c];        // p=0 -> y
    s2[c] = on * off_w[64 + c];   // p=1 -> x
    __syncthreads();
    for (int off = 32; off > 0; off >>= 1) {
        if (c < off) { s1[c] += s1[c + off]; s2[c] += s2[c + off]; }
        __syncthreads();
    }
    if (c == 0) {
        float ry = ((0.5f + (float)i) / 15.0f) * 2.0f - 1.0f;
        float rx = ((0.5f + (float)j) / 15.0f) * 2.0f - 1.0f;
        float py = fminf(fmaxf(s1[0] + ry, -1.0f), 1.0f);
        float px = fminf(fmaxf(s2[0] + rx, -1.0f), 1.0f);
        pos[(bg * 256 + s) * 2 + 0] = py;
        pos[(bg * 256 + s) * 2 + 1] = px;
    }
}

// ---------------------------------------------------------------------------
// Deformed bilinear sampling of x at pos -> x_s (B,256,256)
// ---------------------------------------------------------------------------
__global__ void sample_kernel(const float* __restrict__ x,
                              const float* __restrict__ pos,
                              float* __restrict__ xs) {
    const int idx = blockIdx.x * 256 + threadIdx.x;
    if (idx >= 16 * 64 * 256) return;
    const int s  = idx & 255;
    const int c  = (idx >> 8) & 63;
    const int bg = idx >> 14;
    const int b = bg >> 2, g = bg & 3;

    const float py = pos[(bg * 256 + s) * 2 + 0];
    const float px = pos[(bg * 256 + s) * 2 + 1];
    const float gx = (px + 1.0f) * 0.5f * 63.0f;
    const float gy = (py + 1.0f) * 0.5f * 63.0f;
    const float x0 = floorf(gx), y0 = floorf(gy);
    const float wx = gx - x0, wy = gy - y0;
    const float* src = x + ((size_t)(b * 256 + g * 64 + c)) * 4096;

    auto tap = [&](float xi, float yi) -> float {
        if (xi < 0.0f || xi > 63.0f || yi < 0.0f || yi > 63.0f) return 0.0f;
        return src[(int)yi * 64 + (int)xi];
    };
    float v = tap(x0, y0) * (1.0f - wx) * (1.0f - wy)
            + tap(x0 + 1.0f, y0) * wx * (1.0f - wy)
            + tap(x0, y0 + 1.0f) * (1.0f - wx) * wy
            + tap(x0 + 1.0f, y0 + 1.0f) * wx * wy;
    xs[((size_t)(b * 256 + g * 64 + c)) * 256 + s] = v;
}

// ---------------------------------------------------------------------------
// Fused attention: S = (q^T k)*scale + rpe_bias ; softmax ; O = P * v^T
// grid = (HW/64, B*HEADS), block = 128 (4 wave32, 16 queries/wave).
// LDS 52KB: Vt (16K) | Kt (16K, aliased by Pp 32K after last K use) | Qf (4K)
// All fragment reads are contiguous 2 x ds_load_b128 per lane.
// ---------------------------------------------------------------------------
__global__ void attn_kernel(const float* __restrict__ q,
                            const float* __restrict__ kbuf,
                            const float* __restrict__ vbuf,
                            const float* __restrict__ pos,
                            const float* __restrict__ rpe,
                            float* __restrict__ outbuf) {
    __shared__ __align__(16) _Float16 smem[26624];   // 52 KB
    const int tid  = threadIdx.x;
    const int wave = tid >> 5;
    const int lane = tid & 31;
    const int l16  = lane & 15;
    const int lh   = (lane >> 4) & 1;
    const int qBlock = blockIdx.x * 64;
    const int bh = blockIdx.y;
    const int b = bh >> 3, head = bh & 7;
    const int bg = b * 4 + (head >> 1);

    _Float16* Vt = smem;            // Vt[d*256 + key]                (8192 h)
    _Float16* Kt = smem + 8192;     // Kt[key*32 + c]                 (8192 h)
    _Float16* Pp = smem + 8192;     // Pp[m*256 + kc*32 + perm]       (16384 h)
    _Float16* Qf = smem + 24576;    // Qf[m*32 + a_perm(c)]           (2048 h)

    const float* kp = kbuf + ((size_t)(b * 256 + head * 32)) * 256;
    const float* vp = vbuf + ((size_t)(b * 256 + head * 32)) * 256;
    // K transposed to per-key contiguous channels; V kept per-d contiguous keys
    for (int idx = tid; idx < 32 * 256; idx += 128) {
        int c = idx >> 8, n = idx & 255;
        Kt[n * 32 + c]  = (_Float16)kp[c * 256 + n];
        Vt[c * 256 + n] = (_Float16)vp[c * 256 + n];
    }
    // Q tile -> A-fragment layout (coalesced global read along queries)
    for (int idx = tid; idx < 64 * 32; idx += 128) {
        int m = idx & 63, c = idx >> 6;
        Qf[m * 32 + a_perm(c)] =
            (_Float16)q[((size_t)(b * 256 + head * 32 + c)) * 4096 + qBlock + m];
    }
    __syncthreads();

    const v16h afr = ld_frag(&Qf[(wave * 16 + l16) * 32 + (lh << 4)]);

    // S tiles: 16 queries x 256 keys in 16 WMMAs
    v8f S[16];
    for (int nt = 0; nt < 16; ++nt) {
        for (int j = 0; j < 8; ++j) S[nt][j] = 0.0f;
        const v16h bf = ld_frag(&Kt[(nt * 16 + l16) * 32 + (lh << 4)]);
        S[nt] = __builtin_amdgcn_wmma_f32_16x16x32_f16(
            false, afr, false, bf, (short)0, S[nt], false, false);
    }
    __syncthreads();   // everyone done reading Kt before Pp overwrites it

    // scale + relative-position bias (bilinear on 127x127 bucket table in L2)
    const float scale = 0.17677669529663687f;   // 1/sqrt(32)
    const float* T = rpe + (size_t)head * 127 * 127;
    for (int nt = 0; nt < 16; ++nt) {
        const int key = nt * 16 + l16;
        const float py = pos[(bg * 256 + key) * 2 + 0];
        const float px = pos[(bg * 256 + key) * 2 + 1];
        const float p1y = (py + 1.0f) * 0.5f * 65.0f;
        const float p1x = (px + 1.0f) * 0.5f * 65.0f;
        for (int r = 0; r < 8; ++r) {
            const int qidx = qBlock + wave * 16 + r + (lh << 3);
            const float q1y = (float)(qidx >> 6) * (65.0f / 63.0f);
            const float q1x = (float)(qidx & 63) * (65.0f / 63.0f);
            const float dy = q1y - p1y, dx = q1x - p1x;
            const bool small = (fabsf(dy) + fabsf(dx)) <= 63.0f;
            const float r0 = (small ? dy : 63.0f) * (1.0f / 63.0f) - 1.0f;
            const float r1 = (small ? dx : 63.0f) * (1.0f / 63.0f) - 1.0f;
            const float sx = (r1 + 1.0f) * 63.0f;   // (r+1)/2*(NB-1)
            const float sy = (r0 + 1.0f) * 63.0f;
            const float fx0 = floorf(sx), fy0 = floorf(sy);
            const float wx = sx - fx0, wy = sy - fy0;
            float bias = 0.0f;
            #pragma unroll
            for (int t = 0; t < 4; ++t) {
                float xi = fx0 + (float)(t & 1);
                float yi = fy0 + (float)(t >> 1);
                float w = ((t & 1) ? wx : 1.0f - wx) * ((t >> 1) ? wy : 1.0f - wy);
                if (xi >= 0.0f && xi <= 126.0f && yi >= 0.0f && yi <= 126.0f)
                    bias += T[(int)yi * 127 + (int)xi] * w;
            }
            S[nt][r] = S[nt][r] * scale + bias;
        }
    }

    // softmax per query row; row lives in one half-wave (16 lanes), xor<=8 only
    for (int r = 0; r < 8; ++r) {
        float mx = -3.4e38f;
        for (int nt = 0; nt < 16; ++nt) mx = fmaxf(mx, S[nt][r]);
        for (int off = 8; off >= 1; off >>= 1) mx = fmaxf(mx, __shfl_xor(mx, off, 32));
        float sum = 0.0f;
        for (int nt = 0; nt < 16; ++nt) {
            float e = __expf(S[nt][r] - mx);
            S[nt][r] = e; sum += e;
        }
        for (int off = 8; off >= 1; off >>= 1) sum += __shfl_xor(sum, off, 32);
        const float inv = 1.0f / sum;
        for (int nt = 0; nt < 16; ++nt) S[nt][r] *= inv;
    }

    // stash P into LDS (f16) directly in A-fragment order:
    //   key n = nt*16+l16 -> chunk kc=n>>5, position n&31 (contiguous)
    {
        const int g = (l16 >> 3) & 1;
        const int hbase = l16 & 7;
        for (int nt = 0; nt < 16; ++nt) {
            const int kc = nt >> 1;
            const int h  = hbase | ((nt & 1) << 3);
            const int col = kc * 32 + ((nt & 1) << 4) + l16;
            for (int r = 0; r < 8; ++r) {
                int m = wave * 16 + r + (lh << 3);
                Pp[m * 256 + col] = (_Float16)S[nt][r];
            }
        }
    }
    __syncthreads();

    // O = P(16x256) x V^T(256x32), accumulate over 8 K-chunks of 32
    for (int dt = 0; dt < 2; ++dt) {
        v8f o;
        for (int j = 0; j < 8; ++j) o[j] = 0.0f;
        for (int kc = 0; kc < 8; ++kc) {
            const v16h pa = ld_frag(&Pp[(wave * 16 + l16) * 256 + kc * 32 + (lh << 4)]);
            const v16h vb = ld_frag(&Vt[(dt * 16 + l16) * 256 + kc * 32 + (lh << 4)]);
            o = __builtin_amdgcn_wmma_f32_16x16x32_f16(
                false, pa, false, vb, (short)0, o, false, false);
        }
        for (int r = 0; r < 8; ++r) {
            int m = qBlock + wave * 16 + r + (lh << 3);
            int d = dt * 16 + l16;
            outbuf[((size_t)(b * 256 + head * 32 + d)) * 4096 + m] = o[r];
        }
    }
}

// ---------------------------------------------------------------------------
extern "C" void kernel_launch(void* const* d_in, const int* in_sizes, int n_in,
                              void* d_out, int out_size, void* d_ws, size_t ws_size,
                              hipStream_t stream) {
    const float* x     = (const float*)d_in[0];
    const float* Wq    = (const float*)d_in[1];
    const float* bq    = (const float*)d_in[2];
    const float* Wk    = (const float*)d_in[3];
    const float* bk    = (const float*)d_in[4];
    const float* Wv    = (const float*)d_in[5];
    const float* bv    = (const float*)d_in[6];
    const float* Wo    = (const float*)d_in[7];
    const float* bo    = (const float*)d_in[8];
    const float* dw_w  = (const float*)d_in[9];
    const float* dw_b  = (const float*)d_in[10];
    const float* ln_g  = (const float*)d_in[11];
    const float* ln_b  = (const float*)d_in[12];
    const float* off_w = (const float*)d_in[13];
    const float* rpe   = (const float*)d_in[14];
    float* out = (float*)d_out;

    float* ws      = (float*)d_ws;
    float* q_ws    = ws;                 // 4*256*4096  = 4,194,304 f
    float* attn_ws = ws + 4194304;       // 4,194,304 f
    float* xs_ws   = ws + 8388608;       // 4*256*256   =   262,144 f
    float* k_ws    = ws + 8650752;       //   262,144 f
    float* v_ws    = ws + 8912896;       //   262,144 f
    float* pos_ws  = ws + 9175040;       // 16*256*2    =     8,192 f

    // 1) q projection
    gemm_wmma_kernel<<<dim3(64, 2, 4), 256, 0, stream>>>(Wq, x, bq, q_ws, 256, 4096, 256);
    // 2) offset net -> clipped sampling positions
    offset_kernel<<<dim3(256, 16), 64, 0, stream>>>(q_ws, dw_w, dw_b, ln_g, ln_b, off_w, pos_ws);
    // 3) deformed sampling of x
    sample_kernel<<<dim3(1024), 256, 0, stream>>>(x, pos_ws, xs_ws);
    // 4) k / v projections
    gemm_wmma_kernel<<<dim3(4, 2, 4), 256, 0, stream>>>(Wk, xs_ws, bk, k_ws, 256, 256, 256);
    gemm_wmma_kernel<<<dim3(4, 2, 4), 256, 0, stream>>>(Wv, xs_ws, bv, v_ws, 256, 256, 256);
    // 5) fused attention (scores + rpe bias + softmax + PV)
    attn_kernel<<<dim3(64, 32), 128, 0, stream>>>(q_ws, k_ws, v_ws, pos_ws, rpe, attn_ws);
    // 6) output projection
    gemm_wmma_kernel<<<dim3(64, 2, 4), 256, 0, stream>>>(Wo, attn_ws, bo, out, 256, 4096, 256);
}
